// RNN_47854525612155
// MI455X (gfx1250) — compile-verified, matrix operand order checked by reading, not run on hardware
//
#include <hip/hip_runtime.h>
#include <math.h>

// RNN: out[n][t][h] = h_t,  h_t = tanh(h_{t-1} @ Wh + x_t @ Wx + b)
// N=128, T=512, D=H=1024.  Phase 1 writes xW+b into d_out; phase 2 is a
// persistent-kernel scan that accumulates h@Wh on top of d_out in place.

#define Nb 128
#define Tt 512
#define Dd 1024
#define Hh 1024

typedef __attribute__((ext_vector_type(16))) __bf16 bf16x16;
typedef __attribute__((ext_vector_type(8)))  __bf16 bf16x8;
typedef __attribute__((ext_vector_type(4)))  __bf16 bf16x4;
typedef __attribute__((ext_vector_type(2)))  __bf16 bf16x2;
typedef __attribute__((ext_vector_type(8)))  float  f32x8;
typedef __attribute__((ext_vector_type(4)))  float  f32x4;
typedef __attribute__((ext_vector_type(2)))  float  f32x2;

static __device__ __forceinline__ f32x8 wmma_bf16(bf16x16 a, bf16x16 b, f32x8 c) {
    // D = A(16x32 bf16) * B(32x16 bf16) + C(16x16 f32)
    return __builtin_amdgcn_wmma_f32_16x16x32_bf16(
        /*neg_a=*/false, a, /*neg_b=*/false, b,
        /*c_mod=*/(short)0, c, /*reuse_a=*/false, /*reuse_b=*/false);
}

static __device__ __forceinline__ bf16x16 combine(bf16x8 lo, bf16x8 hi) {
    bf16x16 r;
#pragma unroll
    for (int i = 0; i < 8; ++i) { r[i] = lo[i]; r[i + 8] = hi[i]; }
    return r;
}

// ---------------------------------------------------------------------------
// Phase 1:  out[m][:] = x[m][:] @ Wx + b     (m = n*T + t, row stride H)
// 128x64 output tile per WG; each of 8 waves computes a 2x2 grid of 16x16
// tiles (4 WMMAs / 8 fragment loads per K-step).  K staged in LDS as bf16.
// ---------------------------------------------------------------------------
__global__ void __launch_bounds__(256)
xw_gemm(const float* __restrict__ x, const float* __restrict__ Wx,
        const float* __restrict__ b, float* __restrict__ out) {
    __shared__ __align__(16) __bf16 As[128 * 32];  // A tile, row-major [row][k]
    __shared__ __align__(16) __bf16 Bt[64 * 32];   // B tile transposed [col][k]

    const int tid  = threadIdx.x;
    const int mb   = blockIdx.x * 128;  // 65536 / 128 = 512 blocks in x
    const int cb   = blockIdx.y * 64;   // 1024  / 64  = 16 blocks in y
    const int wave = tid >> 5, lane = tid & 31;
    const int lrow = lane & 15, lhalf = lane >> 4;
    const int rt0  = (wave & 3) * 2,  rt1 = rt0 + 1;   // 8 row tiles, 2/wave
    const int ct0  = (wave >> 2) * 2, ct1 = ct0 + 1;   // 4 col tiles, 2/wave

    f32x8 acc00 = {}, acc01 = {}, acc10 = {}, acc11 = {};

    for (int kt = 0; kt < 32; ++kt) {              // 32 K-steps of 32
        const int kb = kt * 32;
        __syncthreads();
        // Stage A: 128x32 fp32 -> bf16 (1024 float4 loads / 256 threads),
        // packed bf16x4 stores -> ds_store_b64.
#pragma unroll
        for (int i = 0; i < 4; ++i) {
            int id = tid + i * 256;
            int r = id >> 3, c4 = (id & 7) * 4;
            const float* gp = x + (size_t)(mb + r) * Dd + kb + c4;
            f32x4 v = *(const f32x4*)gp;
            if (kt < 31) __builtin_prefetch(gp + 32, 0, 1);  // next K tile
            *(bf16x4*)&As[r * 32 + c4] = __builtin_convertvector(v, bf16x4);
        }
        // Stage B transposed: Bt[n][k] = Wx[kb+k][cb+n]; two K-adjacent
        // elements per thread, packed bf16x2 store -> ds_store_b32.
#pragma unroll
        for (int i = 0; i < 4; ++i) {
            int id = tid + i * 256;
            int n = id & 63, k2 = (id >> 6) * 2;
            f32x2 v;
            v.x = Wx[(size_t)(kb + k2)     * Hh + cb + n];
            v.y = Wx[(size_t)(kb + k2 + 1) * Hh + cb + n];
            *(bf16x2*)&Bt[n * 32 + k2] = __builtin_convertvector(v, bf16x2);
        }
        __syncthreads();

        const int kb8 = lhalf * 8;
        const __bf16* a0p = &As[(rt0 * 16 + lrow) * 32 + kb8];
        const __bf16* a1p = &As[(rt1 * 16 + lrow) * 32 + kb8];
        const __bf16* b0p = &Bt[(ct0 * 16 + lrow) * 32 + kb8];
        const __bf16* b1p = &Bt[(ct1 * 16 + lrow) * 32 + kb8];
        bf16x16 A0 = combine(*(const bf16x8*)a0p, *(const bf16x8*)(a0p + 16));
        bf16x16 A1 = combine(*(const bf16x8*)a1p, *(const bf16x8*)(a1p + 16));
        bf16x16 B0 = combine(*(const bf16x8*)b0p, *(const bf16x8*)(b0p + 16));
        bf16x16 B1 = combine(*(const bf16x8*)b1p, *(const bf16x8*)(b1p + 16));
        acc00 = wmma_bf16(A0, B0, acc00);
        acc01 = wmma_bf16(A0, B1, acc01);
        acc10 = wmma_bf16(A1, B0, acc10);
        acc11 = wmma_bf16(A1, B1, acc11);
    }

    // Epilogue: + bias, store fp32 into d_out (this is xW, pre-scan)
    const int c0 = cb + ct0 * 16 + lrow;
    const int c1 = cb + ct1 * 16 + lrow;
    const float bias0 = b[c0], bias1 = b[c1];
#pragma unroll
    for (int j = 0; j < 8; ++j) {
        int m0 = mb + rt0 * 16 + j + 8 * lhalf;
        int m1 = mb + rt1 * 16 + j + 8 * lhalf;
        out[(size_t)m0 * Hh + c0] = acc00[j] + bias0;
        out[(size_t)m0 * Hh + c1] = acc01[j] + bias1;
        out[(size_t)m1 * Hh + c0] = acc10[j] + bias0;
        out[(size_t)m1 * Hh + c1] = acc11[j] + bias1;
    }
}

// ---------------------------------------------------------------------------
// Phase 2: persistent scan.  64 WGs x 256 threads.  WG owns 16 columns of H
// and keeps Wh^T slab (16x1024 bf16 = 32KB) in LDS for all 512 steps.
// Wave w owns rows 16w..16w+15; per step: 32 chained bf16 WMMAs (2 accums),
// fused +xw / tanh / store into d_out[:,t,:] which is h for step t+1.
// ---------------------------------------------------------------------------
__global__ void __launch_bounds__(256)
rnn_scan(const float* __restrict__ h0, const float* __restrict__ Wh,
         float* __restrict__ out, unsigned* __restrict__ sync) {
    __shared__ __align__(16) __bf16 WhT[16 * 1024];   // [col_local][k]

    const int tid     = threadIdx.x;
    const int colbase = blockIdx.x * 16;              // 64 WGs * 16 = 1024
    for (int idx = tid; idx < 16 * 512; idx += 256) {
        int n = idx & 15, k2 = (idx >> 4) * 2;
        f32x2 v;
        v.x = Wh[(size_t)k2       * Hh + colbase + n];
        v.y = Wh[(size_t)(k2 + 1) * Hh + colbase + n];
        *(bf16x2*)&WhT[n * 1024 + k2] = __builtin_convertvector(v, bf16x2);
    }
    __syncthreads();

    const int wave = tid >> 5, lane = tid & 31;
    const int lrow = lane & 15, lhalf = lane >> 4;
    const int rowbase = wave * 16;                    // 8 waves * 16 = 128 rows
    const int myrow   = rowbase + lrow;               // A row for this lane
    const int col     = colbase + lrow;               // C col for this lane

    unsigned* cnt = sync;
    unsigned* gen = sync + 16;                        // separate cache line

    for (int t = 0; t < Tt; ++t) {
        // h_{t-1}: h0 for t==0, else previous timestep slice of out
        const float* hp = (t == 0)
            ? (h0 + (size_t)myrow * Hh)
            : (out + (size_t)myrow * ((size_t)Tt * Hh) + (size_t)(t - 1) * Hh);

        f32x8 acc0 = {}, acc1 = {};
#pragma unroll 4
        for (int kk = 0; kk < 32; ++kk) {
            const int kb = kk * 32 + lhalf * 8;
            bf16x4 c0 = __builtin_convertvector(*(const f32x4*)(hp + kb),      bf16x4);
            bf16x4 c1 = __builtin_convertvector(*(const f32x4*)(hp + kb + 4),  bf16x4);
            bf16x4 c2 = __builtin_convertvector(*(const f32x4*)(hp + kb + 16), bf16x4);
            bf16x4 c3 = __builtin_convertvector(*(const f32x4*)(hp + kb + 20), bf16x4);
            bf16x16 af;
#pragma unroll
            for (int i = 0; i < 4; ++i) {
                af[i]      = c0[i];
                af[4 + i]  = c1[i];
                af[8 + i]  = c2[i];
                af[12 + i] = c3[i];
            }
            const __bf16* bp = &WhT[lrow * 1024 + kk * 32 + lhalf * 8];
            bf16x16 bfv = combine(*(const bf16x8*)bp, *(const bf16x8*)(bp + 16));
            if (kk & 1) acc1 = wmma_bf16(af, bfv, acc1);
            else        acc0 = wmma_bf16(af, bfv, acc0);
        }

        // Fused epilogue: h_t = tanh(h@Wh + xw_t), in place over d_out
#pragma unroll
        for (int j = 0; j < 8; ++j) {
            int m = rowbase + j + 8 * lhalf;
            size_t o = (size_t)m * ((size_t)Tt * Hh) + (size_t)t * Hh + col;
            float v = acc0[j] + acc1[j] + out[o];
            out[o] = tanhf(v);
        }

        // Grid-wide barrier: all h_t written before anyone reads it at t+1
        __syncthreads();
        __threadfence();                               // release
        if (tid == 0) {
            if (atomicAdd(cnt, 1u) == gridDim.x - 1) {
                atomicExch(cnt, 0u);
                atomicAdd(gen, 1u);
            } else {
                while (atomicAdd(gen, 0u) <= (unsigned)t)
                    __builtin_amdgcn_s_sleep(2);
            }
        }
        __syncthreads();
        __threadfence();                               // acquire
    }
}

extern "C" void kernel_launch(void* const* d_in, const int* in_sizes, int n_in,
                              void* d_out, int out_size, void* d_ws, size_t ws_size,
                              hipStream_t stream) {
    (void)in_sizes; (void)n_in; (void)out_size; (void)ws_size;
    const float* x  = (const float*)d_in[0];   // (N, T, D)
    const float* h0 = (const float*)d_in[1];   // (N, H)
    const float* Wx = (const float*)d_in[2];   // (D, H)
    const float* Wh = (const float*)d_in[3];   // (H, H)
    const float* b  = (const float*)d_in[4];   // (H,)
    float* out = (float*)d_out;                // (N, T, H)

    hipMemsetAsync(d_ws, 0, 256, stream);      // barrier counters

    dim3 g1((Nb * Tt) / 128, Hh / 64);         // 512 x 16 workgroups
    xw_gemm<<<g1, 256, 0, stream>>>(x, Wx, b, out);

    rnn_scan<<<64, 256, 0, stream>>>(h0, Wh, out, (unsigned*)d_ws);
}
